// MultiHeadAttention_14310831030707
// MI455X (gfx1250) — matrix operand order repaired and checked by reading, not compiled, not run on hardware
//
#include <hip/hip_runtime.h>

#define DMODEL 1024
#define NHEAD 16
#define HEADD 64
#define BATCH 2
#define SEQ 2048

typedef __attribute__((ext_vector_type(16))) __bf16 v16bf;
typedef __attribute__((ext_vector_type(8)))  float  v8f;

union BF16Frag { unsigned u[8]; v16bf v; };

// float -> bf16, round-to-nearest-even, branch-free
__device__ __forceinline__ unsigned short f2bf(float x) {
  unsigned u = __float_as_uint(x);
  unsigned r = 0x7FFFu + ((u >> 16) & 1u);
  return (unsigned short)((u + r) >> 16);
}
__device__ __forceinline__ unsigned pack2(float lo, float hi) {
  return (unsigned)f2bf(lo) | ((unsigned)f2bf(hi) << 16);
}

// A-operand (16x32 bf16) K index for vgpr v, lane-half h:
// lanes 0-15: K = 0..7,16..23 ; lanes 16-31: K = 8..15,24..31
__device__ __forceinline__ int kmapA(int v, int h) {
  return ((v < 4) ? (2 * v) : (8 + 2 * v)) + 8 * h;
}

// Load a 16x32 A-fragment from row-major bf16 (two b128 loads).
__device__ __forceinline__ void loadA(BF16Frag& f, const unsigned short* row,
                                      int half) {
  uint4 t0 = *(const uint4*)(row + 8 * half);        // K 0..7  (+8 for half 1)
  uint4 t1 = *(const uint4*)(row + 16 + 8 * half);   // K 16..23
  f.u[0] = t0.x; f.u[1] = t0.y; f.u[2] = t0.z; f.u[3] = t0.w;
  f.u[4] = t1.x; f.u[5] = t1.y; f.u[6] = t1.z; f.u[7] = t1.w;
}
// Load a 32x16 B-fragment: lane = N column, K 0..7,16..23 / 8..15,24..31 by half.
// Works for both global and LDS source pointers.
__device__ __forceinline__ void loadB(BF16Frag& f, const unsigned short* row,
                                      int half) {
  uint4 t0 = *(const uint4*)(row + 8 * half);
  uint4 t1 = *(const uint4*)(row + 16 + 8 * half);
  f.u[0] = t0.x; f.u[1] = t0.y; f.u[2] = t0.z; f.u[3] = t0.w;
  f.u[4] = t1.x; f.u[5] = t1.y; f.u[6] = t1.z; f.u[7] = t1.w;
}

// -------- fp32 -> bf16 bulk conversion (bandwidth bound, one pass) --------
__global__ __launch_bounds__(256) void cvt_f32_bf16(
    const float* __restrict__ in, unsigned short* __restrict__ out, int n4) {
  int i = blockIdx.x * blockDim.x + threadIdx.x;
  if (i < n4) {
    float4 d = ((const float4*)in)[i];
    uint2 p;
    p.x = pack2(d.x, d.y);
    p.y = pack2(d.z, d.w);
    ((uint2*)out)[i] = p;
  }
}

// -------- bit-pack the attention mask: 1 bit per key via wave32 ballot -----
__global__ __launch_bounds__(256) void pack_mask(const int* __restrict__ mask,
                                                 unsigned* __restrict__ bits) {
  int i = blockIdx.x * blockDim.x + threadIdx.x;
  unsigned long long bal = __ballot(mask[i] != 0);
  if ((threadIdx.x & 31) == 0) bits[i >> 5] = (unsigned)bal;
}

// C = A[M,K] @ W[N,K]^T + bias; A,W bf16. 64x64 output block per wave
// (4x4 WMMA tiles): 16 WMMAs per 16 b128 loads per K-step.
// MODE 0: store bf16, head-split [B,H,S,Dh]
// MODE 1: store f32, plain row-major [M,N]
// MODE 2: store bf16, V-transposed [B,H,Dh,S]
template <int MODE>
__global__ __launch_bounds__(256) void gemm64(
    const unsigned short* __restrict__ A, const unsigned short* __restrict__ W,
    const float* __restrict__ bias, void* __restrict__ out, int M, int N,
    int K) {
  const int lane = threadIdx.x & 31;
  const int wid  = blockIdx.x * 8 + (threadIdx.x >> 5);
  const int TN   = N >> 6;
  const int bm = wid / TN, bn = wid % TN;
  if (bm * 64 >= M) return;
  const int half = lane >> 4, mr = lane & 15;

  const unsigned short* abase = A + (size_t)(bm * 64 + mr) * K;
  const unsigned short* wbase = W + (size_t)(bn * 64 + mr) * K;
  const size_t rstep = (size_t)16 * K;

  v8f zero = {};
  v8f acc[4][4];
#pragma unroll
  for (int i = 0; i < 4; ++i)
#pragma unroll
    for (int j = 0; j < 4; ++j) acc[i][j] = zero;

  for (int k0 = 0; k0 < K; k0 += 32) {
    BF16Frag af[4], bf[4];
#pragma unroll
    for (int i = 0; i < 4; ++i) loadA(af[i], abase + i * rstep + k0, half);
#pragma unroll
    for (int j = 0; j < 4; ++j) loadB(bf[j], wbase + j * rstep + k0, half);
    if (k0 + 128 < K) {  // emits global_prefetch_b8
#pragma unroll
      for (int i = 0; i < 4; ++i)
        __builtin_prefetch(abase + i * rstep + k0 + 128, 0, 1);
    }
#pragma unroll
    for (int i = 0; i < 4; ++i)
#pragma unroll
      for (int j = 0; j < 4; ++j)
        acc[i][j] = __builtin_amdgcn_wmma_f32_16x16x32_bf16(
            false, af[i].v, false, bf[j].v, (short)0, acc[i][j], false, false);
  }

  const int c = lane & 15;
#pragma unroll
  for (int j = 0; j < 4; ++j) {
    int n = bn * 64 + j * 16 + c;
    float bv = bias[n];
#pragma unroll
    for (int i = 0; i < 4; ++i) {
#pragma unroll
      for (int r = 0; r < 8; ++r) {
        int m = bm * 64 + i * 16 + r + 8 * half;
        float val = acc[i][j][r] + bv;
        if (MODE == 1) {
          ((float*)out)[(size_t)m * N + n] = val;
        } else {
          int b = m / SEQ, s = m % SEQ;
          int h = n / HEADD, d = n % HEADD;
          size_t idx = (MODE == 0)
              ? (((size_t)(b * NHEAD + h) * SEQ + s) * HEADD + d)
              : (((size_t)(b * NHEAD + h) * HEADD + d) * SEQ + s);
          ((unsigned short*)out)[idx] = f2bf(val);
        }
      }
    }
  }
}

// Flash attention, block-cooperative: 8 waves of a block share one (b,h) and
// stage each 64-key K / V^T chunk into LDS once (8 KB each) using the gfx1250
// async global->LDS engine (ASYNCcnt), cutting global K/V traffic 8x.
// Each wave owns a 16-query tile; 64 keys per iteration (16 WMMAs) with
// online softmax; P goes through LDS as bf16 to convert C-layout ->
// A-operand layout for the PV matmuls.
__global__ __launch_bounds__(256) void attn_kernel(
    const unsigned short* __restrict__ Qs,  // [B*H, S, 64] bf16
    const unsigned short* __restrict__ Ks,  // [B*H, S, 64] bf16
    const unsigned short* __restrict__ Vt,  // [B*H, 64, S] bf16 (transposed)
    const unsigned* __restrict__ mbits,     // [S, S/32] packed mask bits
    unsigned short* __restrict__ out) {     // [B*S, 1024] bf16
  __shared__ __align__(16) unsigned short kst[64][72];      // keys x dh
  __shared__ __align__(16) unsigned short vst[64][72];      // dh x keys
  __shared__ __align__(16) unsigned short plds[8][16][72];  // P transpose

  const int tid  = threadIdx.x;
  const int lane = tid & 31;
  const int w    = tid >> 5;
  const int bh   = blockIdx.x >> 4;       // 32 (b,h) pairs
  const int qg   = blockIdx.x & 15;       // query group of 8 tiles
  const int qt   = qg * 8 + w;            // this wave's 16-query tile
  const int b = bh / NHEAD, h = bh % NHEAD;
  const int half = lane >> 4, mr = lane & 15;
  const int qrow0 = qt * 16;
  const int c = lane & 15;
  const unsigned bitlo = 1u << c;
  const unsigned bithi = bitlo << 16;

  // staging coordinates: each thread copies 2 x 16B rows-chunks per buffer
  const int srow = tid >> 3;              // 0..31
  const int scol = (tid & 7) * 8;         // 0..56 shorts (16B granules)
  const unsigned short* kgbase = Ks + (size_t)bh * SEQ * HEADD;
  const unsigned short* vgbase = Vt + (size_t)bh * HEADD * SEQ;

  // Q A-fragments for Dh halves 0..31 and 32..63
  const unsigned short* qbase = Qs + ((size_t)bh * SEQ + qrow0 + mr) * HEADD;
  BF16Frag qa0, qa1;
  loadA(qa0, qbase, half);
  loadA(qa1, qbase + 32, half);

  float rowmax[8], rowsum[8];
  v8f zero = {};
  v8f oacc[4];
#pragma unroll
  for (int r = 0; r < 8; ++r) { rowmax[r] = -3.0e38f; rowsum[r] = 0.0f; }
#pragma unroll
  for (int g = 0; g < 4; ++g) oacc[g] = zero;

  const float scale = 0.125f;  // 1/sqrt(64)

  for (int kv = 0; kv < SEQ; kv += 64) {
    // ---- async stage of K[kv..kv+63][:] and V^T[:][kv..kv+63] into LDS ----
    // Each lane copies 16B global->LDS directly (no VGPR round-trip),
    // tracked by ASYNCcnt.
#pragma unroll
    for (int pp = 0; pp < 2; ++pp) {
      int rr = srow + 32 * pp;
      unsigned long long kg =
          (unsigned long long)(size_t)(kgbase + (size_t)(kv + rr) * HEADD +
                                       scol);
      unsigned long long vg =
          (unsigned long long)(size_t)(vgbase + (size_t)rr * SEQ + kv + scol);
      unsigned kl = (unsigned)(size_t)&kst[rr][scol];  // LDS byte address
      unsigned vl = (unsigned)(size_t)&vst[rr][scol];
      asm volatile("global_load_async_to_lds_b128 %0, %1, off"
                   :
                   : "v"(kl), "v"(kg)
                   : "memory");
      asm volatile("global_load_async_to_lds_b128 %0, %1, off"
                   :
                   : "v"(vl), "v"(vg)
                   : "memory");
    }
    if (kv + 64 < SEQ) {  // prefetch next chunk (global_prefetch_b8)
      __builtin_prefetch(kgbase + (size_t)(kv + 64 + srow) * HEADD + scol, 0, 1);
      __builtin_prefetch(vgbase + (size_t)srow * SEQ + kv + 64 + scol, 0, 1);
    }
    asm volatile("s_wait_asynccnt 0x0" ::: "memory");
    __syncthreads();

    // ---- scores for 64 keys: 4 tiles of 16 keys, K frags from LDS ----
    v8f s[4];
#pragma unroll
    for (int t = 0; t < 4; ++t) {
      const unsigned short* kb = &kst[16 * t + mr][0];
      BF16Frag f0, f1;
      loadB(f0, kb, half);        // dh 0..31
      loadB(f1, kb + 32, half);   // dh 32..63
      v8f st = __builtin_amdgcn_wmma_f32_16x16x32_bf16(
          false, qa0.v, false, f0.v, (short)0, zero, false, false);
      s[t] = __builtin_amdgcn_wmma_f32_16x16x32_bf16(
          false, qa1.v, false, f1.v, (short)0, st, false, false);
    }

    // ---- scale + mask(bits) + online softmax ----
#pragma unroll
    for (int r = 0; r < 8; ++r) {
      int qrow = qrow0 + r + 8 * half;
      uint2 mb = *(const uint2*)(mbits + (size_t)qrow * (SEQ / 32) +
                                 (kv >> 5));  // 64 mask bits for this row
      float sv[4];
      sv[0] = (mb.x & bitlo) ? s[0][r] * scale : -10000.0f;
      sv[1] = (mb.x & bithi) ? s[1][r] * scale : -10000.0f;
      sv[2] = (mb.y & bitlo) ? s[2][r] * scale : -10000.0f;
      sv[3] = (mb.y & bithi) ? s[3][r] * scale : -10000.0f;

      float tmax = fmaxf(fmaxf(sv[0], sv[1]), fmaxf(sv[2], sv[3]));
#pragma unroll
      for (int off = 1; off < 16; off <<= 1)
        tmax = fmaxf(tmax, __shfl_xor(tmax, off, 32));
      float mnew  = fmaxf(rowmax[r], tmax);
      float fcorr = __expf(rowmax[r] - mnew);
      float p[4];
#pragma unroll
      for (int t = 0; t < 4; ++t) p[t] = __expf(sv[t] - mnew);
      float psum = (p[0] + p[1]) + (p[2] + p[3]);
#pragma unroll
      for (int off = 1; off < 16; off <<= 1)
        psum += __shfl_xor(psum, off, 32);
      rowsum[r] = rowsum[r] * fcorr + psum;
      rowmax[r] = mnew;
#pragma unroll
      for (int g = 0; g < 4; ++g) oacc[g][r] *= fcorr;

      int m = r + 8 * half;
#pragma unroll
      for (int t = 0; t < 4; ++t) plds[w][m][16 * t + c] = f2bf(p[t]);
    }
    __syncthreads();

    // ---- re-read P (bf16) in A-operand layout: packed dword loads ----
    BF16Frag pa0, pa1;
#pragma unroll
    for (int v = 0; v < 8; ++v) {
      int kA = kmapA(v, half);
      pa0.u[v] = *(const unsigned*)&plds[w][mr][kA];        // keys 0..31
      pa1.u[v] = *(const unsigned*)&plds[w][mr][32 + kA];   // keys 32..63
    }

    // ---- O += P @ V ; V frags from LDS ----
#pragma unroll
    for (int g = 0; g < 4; ++g) {
      const unsigned short* vb = &vst[g * 16 + mr][0];
      BF16Frag v0, v1;
      loadB(v0, vb, half);        // keys 0..31
      loadB(v1, vb + 32, half);   // keys 32..63
      oacc[g] = __builtin_amdgcn_wmma_f32_16x16x32_bf16(
          false, pa0.v, false, v0.v, (short)0, oacc[g], false, false);
      oacc[g] = __builtin_amdgcn_wmma_f32_16x16x32_bf16(
          false, pa1.v, false, v1.v, (short)0, oacc[g], false, false);
    }
    __syncthreads();  // protect kst/vst/plds before next iteration's stage
  }

  // ---- epilogue: normalize, write bf16 [B,S,H*Dh] for output projection ----
#pragma unroll
  for (int r = 0; r < 8; ++r) {
    float inv = 1.0f / rowsum[r];
    int qrow = qrow0 + r + 8 * half;
#pragma unroll
    for (int g = 0; g < 4; ++g) {
      int d = g * 16 + c;
      out[((size_t)b * SEQ + qrow) * DMODEL + h * HEADD + d] =
          f2bf(oacc[g][r] * inv);
    }
  }
}

extern "C" void kernel_launch(void* const* d_in, const int* in_sizes, int n_in,
                              void* d_out, int out_size, void* d_ws,
                              size_t ws_size, hipStream_t stream) {
  (void)in_sizes; (void)n_in; (void)out_size; (void)ws_size;
  const float* q    = (const float*)d_in[0];
  const float* k    = (const float*)d_in[1];
  const float* v    = (const float*)d_in[2];
  const int*   mask = (const int*)d_in[3];
  const float* wq = (const float*)d_in[4];
  const float* bq = (const float*)d_in[5];
  const float* wk = (const float*)d_in[6];
  const float* bk = (const float*)d_in[7];
  const float* wv = (const float*)d_in[8];
  const float* bv = (const float*)d_in[9];
  const float* wo = (const float*)d_in[10];
  const float* bo = (const float*)d_in[11];

  const int M = BATCH * SEQ;                      // 4096
  const size_t XE = (size_t)M * DMODEL;           // 4 Mi elements
  const size_t WE = (size_t)DMODEL * DMODEL;      // 1 Mi elements

  unsigned short* xb    = (unsigned short*)d_ws;  // staging: input bf16, 8 MiB
  unsigned short* wb    = xb + XE;                // staging: weight bf16, 2 MiB
  unsigned short* qs    = wb + WE;                // [B,H,S,Dh]  8 MiB
  unsigned short* ks    = qs + XE;                // [B,H,S,Dh]  8 MiB
  unsigned short* vt    = ks + XE;                // [B,H,Dh,S]  8 MiB
  unsigned short* attnb = vt + XE;                // [B,S,D]     8 MiB
  unsigned*       mbits = (unsigned*)(attnb + XE);  // [S, S/32]  512 KiB

  dim3 blk(256);
  const int cvtx_blocks = (int)(XE / 4 / 256);    // 4096
  const int cvtw_blocks = (int)(WE / 4 / 256);    // 1024
  const int gemm_blocks = (M / 64) * (DMODEL / 64) / 8;      // 128
  const int attn_blocks = (BATCH * NHEAD) * 16;              // 512
  const int mask_blocks = (SEQ * SEQ) / 256;                 // 16384

  // mask bit-pack (independent of projections)
  pack_mask<<<mask_blocks, blk, 0, stream>>>(mask, mbits);

  // Q projection
  cvt_f32_bf16<<<cvtx_blocks, blk, 0, stream>>>(q, xb, (int)(XE / 4));
  cvt_f32_bf16<<<cvtw_blocks, blk, 0, stream>>>(wq, wb, (int)(WE / 4));
  gemm64<0><<<gemm_blocks, blk, 0, stream>>>(xb, wb, bq, qs, M, DMODEL, DMODEL);
  // K projection
  cvt_f32_bf16<<<cvtx_blocks, blk, 0, stream>>>(k, xb, (int)(XE / 4));
  cvt_f32_bf16<<<cvtw_blocks, blk, 0, stream>>>(wk, wb, (int)(WE / 4));
  gemm64<0><<<gemm_blocks, blk, 0, stream>>>(xb, wb, bk, ks, M, DMODEL, DMODEL);
  // V projection (stored transposed per head)
  cvt_f32_bf16<<<cvtx_blocks, blk, 0, stream>>>(v, xb, (int)(XE / 4));
  cvt_f32_bf16<<<cvtw_blocks, blk, 0, stream>>>(wv, wb, (int)(WE / 4));
  gemm64<2><<<gemm_blocks, blk, 0, stream>>>(xb, wb, bv, vt, M, DMODEL, DMODEL);

  // attention (writes bf16, ready for output projection)
  attn_kernel<<<attn_blocks, blk, 0, stream>>>(qs, ks, vt, mbits, attnb);

  // output projection -> f32 result
  cvt_f32_bf16<<<cvtw_blocks, blk, 0, stream>>>(wo, wb, (int)(WE / 4));
  gemm64<1><<<gemm_blocks, blk, 0, stream>>>(attnb, wb, bo, d_out, M, DMODEL,
                                             DMODEL);
}
